// HGTModel_2302102471074
// MI455X (gfx1250) — matrix operand order repaired and checked by reading, not compiled, not run on hardware
//
#include <hip/hip_runtime.h>
#include <hip/hip_bf16.h>
#include <math.h>

// ---------------------------------------------------------------------------
// HGT forward for MI455X (gfx1250, wave32).
//  - dense matrix work: v_wmma_f32_16x16x32_bf16 (f32 accum)
//  - GEMM A-tile staging: TDM tensor_load_to_lds + s_wait_tensorcnt
//  - weights pre-swizzled into WMMA B-fragment order (vector loads, no repack)
//  - edge softmax/aggregation: bf16 gathers + fp32 hardware atomics
// ---------------------------------------------------------------------------

typedef __attribute__((ext_vector_type(16))) __bf16 bf16x16;
typedef __attribute__((ext_vector_type(8)))  float  f32x8;
typedef unsigned int u32x4 __attribute__((ext_vector_type(4)));
typedef int          i32x4t __attribute__((ext_vector_type(4)));
typedef int          i32x8t __attribute__((ext_vector_type(8)));

#define HID 256
#define HEADS 8
#define DD 32

// ---- float <-> orderable-uint encoding for atomic segment-max --------------
__device__ __forceinline__ unsigned ford(float f) {
  unsigned u = __float_as_uint(f);
  return (u & 0x80000000u) ? ~u : (u | 0x80000000u);
}
__device__ __forceinline__ float forddec(unsigned o) {
  return __uint_as_float((o & 0x80000000u) ? (o & 0x7FFFFFFFu) : ~o);
}

// ---- TDM: async 2D tile load (global -> LDS), D# per CDNA5 ISA ch.8 --------
__device__ __forceinline__ void tdm_load_tile_2d(unsigned lds_off, const void* gptr,
                                                 unsigned tile_w, unsigned tile_h,
                                                 unsigned stride_elems) {
  unsigned long long ga = (unsigned long long)(size_t)gptr;
  u32x4 g0;
  g0.x = 1u;                                                   // count=1 (user D#)
  g0.y = lds_off;                                              // LDS byte address
  g0.z = (unsigned)ga;                                         // global addr [31:0]
  g0.w = (unsigned)((ga >> 32) & 0x01FFFFFFu) | 0x80000000u;   // addr[56:32] | type=2
  i32x8t g1;
  g1[0] = 0x00010000;                                          // wg_mask=0, data_size=1 (2B)
  g1[1] = (int)((tile_w & 0xFFFFu) << 16);                     // tensor_dim0 lo -> [31:16]
  g1[2] = (int)(((tile_w >> 16) & 0xFFFFu) | ((tile_h & 0xFFFFu) << 16)); // dim0 hi | dim1 lo
  g1[3] = (int)(((tile_h >> 16) & 0xFFFFu) | ((tile_w & 0xFFFFu) << 16)); // dim1 hi | tile_dim0
  g1[4] = (int)(tile_h & 0xFFFFu);                             // tile_dim1 (tile_dim2 = 0)
  g1[5] = (int)stride_elems;                                   // tensor_dim0_stride lo32
  g1[6] = 0;                                                   // stride hi | dim1_stride lo
  g1[7] = 0;
  i32x4t z4 = {0, 0, 0, 0};
#if defined(__clang_major__) && (__clang_major__ >= 23)
  i32x8t z8 = {0, 0, 0, 0, 0, 0, 0, 0};
  __builtin_amdgcn_tensor_load_to_lds(g0, g1, z4, z4, z8, 0);
#else
  __builtin_amdgcn_tensor_load_to_lds(g0, g1, z4, z4, 0);
#endif
}

// ---- A fragment (16-bit A 16x32, ISA 7.12.2) via u32 quads -> b128 loads ---
// lanes 0-15: row M=lane, K 0..7 (v0-3) and 16..23 (v4-7);
// lanes 16-31: row M=lane-16, K 8..15 and 24..31.  K-pairs are adjacent u32s.
__device__ __forceinline__ bf16x16 load_a_frag_u32(const __bf16* A, int lda32) {
  int lane = threadIdx.x & 31;
  int m = lane & 15, half = lane >> 4;
  const unsigned* A32 = (const unsigned*)A;
  union { bf16x16 v; unsigned u[8]; } r;
#pragma unroll
  for (int j = 0; j < 4; ++j) r.u[j] = A32[m * lda32 + half * 4 + j];
#pragma unroll
  for (int j = 0; j < 4; ++j) r.u[4 + j] = A32[m * lda32 + 8 + half * 4 + j];
  return r.v;
}

// ---------------------------------------------------------------------------
// Weight swizzle: f32 row-major [batch, K, N] -> bf16 WMMA-B-fragment order.
// Per 32x16 tile: 32 lanes x 16 bf16 contiguous (lane = half*16 + n;
// element 2j/2j+1 = B[k][n], B[k+1][n], k = k0+half*8+2j+(j>=4?8:0)).
// ---------------------------------------------------------------------------
__global__ void swizzle_b_kernel(const float* __restrict__ W, __bf16* __restrict__ out,
                                 int K, int N, int batch) {
  int t = blockIdx.x * blockDim.x + threadIdx.x;
  int tilesPer = (K >> 5) * (N >> 4);
  int total = batch * tilesPer * 32;
  if (t >= total) return;
  int lane = t & 31;
  int tile = (t >> 5) % tilesPer;
  int b    = (t >> 5) / tilesPer;
  int ntn = N >> 4;
  int kt = tile / ntn, nt = tile % ntn;
  int n = lane & 15, half = lane >> 4;
  const float* Wb = W + (size_t)b * K * N;
  __bf16* o = out + (size_t)b * K * N + (size_t)tile * 512 + (size_t)lane * 16;
#pragma unroll
  for (int j = 0; j < 8; ++j) {
    int k = kt * 32 + half * 8 + 2 * j + (j >= 4 ? 8 : 0);
    o[2 * j]     = (__bf16)Wb[(size_t)k * N + nt * 16 + n];
    o[2 * j + 1] = (__bf16)Wb[(size_t)(k + 1) * N + nt * 16 + n];
  }
}

// ---------------------------------------------------------------------------
// Generic GEMM: C[M,256] = A[M,K] @ W[K,256] + bias, optional fused epilogue:
//   out = relu(s*o + (1-s)*Xold)  (s = sigmoid(*skipPtr)), written in-place.
// Block: 128 threads (4 waves), block tile 16x64; A tile staged via TDM.
// W is frag-packed (see swizzle_b_kernel).
// ---------------------------------------------------------------------------
__global__ void __launch_bounds__(128)
gemm_bf16_kernel(const __bf16* __restrict__ A, int lda,
                 const __bf16* __restrict__ W,
                 const float* __restrict__ bias,
                 float* Cf, __bf16* Cb,
                 int M, int N, int K,
                 const float* Xold, const float* skipPtr) {
  __shared__ __attribute__((aligned(16))) __bf16 As[16 * 32];
  const int wid  = threadIdx.x >> 5;
  const int lane = threadIdx.x & 31;
  const int m0 = blockIdx.x * 16;
  const int n0 = blockIdx.y * 64 + wid * 16;
  const unsigned ldsAs = (unsigned)(size_t)(void*)As;  // LDS aperture: low 32 bits
  f32x8 acc = {};
  for (int k0 = 0; k0 < K; k0 += 32) {
    __syncthreads();
    if (wid == 0) {
      tdm_load_tile_2d(ldsAs, A + (size_t)m0 * lda + k0, 32u, 16u, (unsigned)lda);
      __builtin_amdgcn_s_wait_tensorcnt(0);
    }
    __syncthreads();
    bf16x16 afrag = load_a_frag_u32(As, 16);
    bf16x16 bfrag = *(const bf16x16*)(W +
        ((size_t)(k0 >> 5) * (size_t)(N >> 4) + (size_t)(n0 >> 4)) * 512 +
        (size_t)lane * 16);
    acc = __builtin_amdgcn_wmma_f32_16x16x32_bf16(false, afrag, false, bfrag,
                                                  (short)0, acc, false, false);
  }
  const int nn = lane & 15;
  const int mh = (lane >> 4) * 8;
  float s = 0.f, onems = 0.f;
  if (skipPtr) { s = 1.f / (1.f + __expf(-skipPtr[0])); onems = 1.f - s; }
#pragma unroll
  for (int j = 0; j < 8; ++j) {
    int row = m0 + j + mh;
    int col = n0 + nn;
    float v = acc[j] + (bias ? bias[col] : 0.f);
    if (skipPtr) {
      v = s * v + onems * Xold[(size_t)row * N + col];
      v = fmaxf(v, 0.f);   // ReLU between layers / before classifier
    }
    if (Cf) Cf[(size_t)row * N + col] = v;
    if (Cb) Cb[(size_t)row * N + col] = (__bf16)v;
  }
}

// ---------------------------------------------------------------------------
// Per-head relation transform: Y[n,h,:] = X[n,h,:] @ R[h] (32x32), bf16 WMMA.
// One wave per (16-node tile, head); 2 WMMAs (output cols split 16+16).
// R is frag-packed per head (2 tiles of 512 elems).
// ---------------------------------------------------------------------------
__global__ void __launch_bounds__(128)
rel_transform_kernel(const __bf16* __restrict__ X, const __bf16* __restrict__ R,
                     __bf16* __restrict__ Y, int nNodes) {
  int wg   = blockIdx.x * 4 + (threadIdx.x >> 5);
  int lane = threadIdx.x & 31;
  int tile = wg >> 3, head = wg & 7;
  if (tile * 16 >= nNodes) return;
  const __bf16* Xp = X + (size_t)tile * 16 * HID + head * DD;  // lda = 256
  bf16x16 afrag = load_a_frag_u32(Xp, HID / 2);
#pragma unroll
  for (int hn = 0; hn < 2; ++hn) {
    bf16x16 bfrag = *(const bf16x16*)(R + (size_t)head * 1024 +
                                      (size_t)hn * 512 + (size_t)lane * 16);
    f32x8 acc = {};
    acc = __builtin_amdgcn_wmma_f32_16x16x32_bf16(false, afrag, false, bfrag,
                                                  (short)0, acc, false, false);
    int nn = lane & 15, mh = (lane >> 4) * 8;
#pragma unroll
    for (int j = 0; j < 8; ++j) {
      int row = tile * 16 + j + mh;
      Y[(size_t)row * HID + head * DD + hn * 16 + nn] = (__bf16)acc[j];
    }
  }
}

// ---------------------------------------------------------------------------
// Edge kernels (one thread per (edge, head))
// ---------------------------------------------------------------------------
__global__ void edge_alpha_kernel(const int* __restrict__ src, const int* __restrict__ dst,
                                  const __bf16* __restrict__ Q, const __bf16* __restrict__ K,
                                  const float* __restrict__ p_rel,
                                  float* __restrict__ alpha, unsigned* amax, int n) {
  int t = blockIdx.x * blockDim.x + threadIdx.x;
  if (t >= n) return;
  int e = t >> 3, h = t & 7;
  int s = src[e], d = dst[e];
  const __bf16* q = Q + (size_t)d * HID + h * DD;
  const __bf16* k = K + (size_t)s * HID + h * DD;
  float acc = 0.f;
#pragma unroll
  for (int i = 0; i < DD; ++i) acc += (float)q[i] * (float)k[i];
  acc *= p_rel[h] * 0.17677669529663687f;  // 1/sqrt(32)
  alpha[t] = acc;
  atomicMax(&amax[(size_t)d * HEADS + h], ford(acc));
}

__global__ void amax_fin_kernel(unsigned* a, int n) {
  int t = blockIdx.x * blockDim.x + threadIdx.x;
  if (t >= n) return;
  float f = forddec(a[t]);
  if (!__builtin_isfinite(f)) f = 0.f;
  ((float*)a)[t] = f;
}

__global__ void edge_expsum_kernel(const int* __restrict__ dst, float* alpha,
                                   const float* __restrict__ amaxf, float* denom, int n) {
  int t = blockIdx.x * blockDim.x + threadIdx.x;
  if (t >= n) return;
  int e = t >> 3, h = t & 7;
  int d = dst[e];
  float ea = __expf(alpha[t] - amaxf[(size_t)d * HEADS + h]);
  alpha[t] = ea;
  unsafeAtomicAdd(&denom[(size_t)d * HEADS + h], ea);
}

__global__ void edge_scatter_kernel(const int* __restrict__ src, const int* __restrict__ dst,
                                    const float* __restrict__ ea, const __bf16* __restrict__ V,
                                    float* scat, int n) {
  int t = blockIdx.x * blockDim.x + threadIdx.x;
  if (t >= n) return;
  int e = t >> 3, h = t & 7;
  int s = src[e], d = dst[e];
  float w = ea[t];
  const __bf16* v = V + (size_t)s * HID + h * DD;
  float* o = scat + (size_t)d * HID + h * DD;
#pragma unroll
  for (int i = 0; i < DD; ++i) unsafeAtomicAdd(&o[i], w * (float)v[i]);
}

// normalize per-relation partial sums, exact GELU, emit bf16 for output GEMM
__global__ void combine_gelu_kernel(const float* __restrict__ s0, const float* __restrict__ d0,
                                    const float* s1, const float* d1,
                                    __bf16* __restrict__ g, int n) {
  int t = blockIdx.x * blockDim.x + threadIdx.x;
  if (t >= n) return;
  int node = t >> 8, c = t & 255, h = c >> 5;
  float v = s0[t] / (d0[(size_t)node * HEADS + h] + 1e-16f);
  if (s1) v += s1[t] / (d1[(size_t)node * HEADS + h] + 1e-16f);
  float gl = 0.5f * v * (1.0f + erff(v * 0.70710678118654752f));
  g[t] = (__bf16)gl;
}

// ---------------------------------------------------------------------------
// Misc small kernels
// ---------------------------------------------------------------------------
__global__ void f32_to_bf16_kernel(const float* __restrict__ a, __bf16* __restrict__ b, int n) {
  int t = blockIdx.x * blockDim.x + threadIdx.x;
  if (t < n) b[t] = (__bf16)a[t];
}
__global__ void fill_u32_kernel(unsigned* p, unsigned v, int n) {
  int t = blockIdx.x * blockDim.x + threadIdx.x;
  if (t < n) p[t] = v;
}
__global__ void gather_embed_kernel(const int* __restrict__ ids, const float* __restrict__ emb,
                                    int embRows, float* __restrict__ xf, __bf16* __restrict__ xb,
                                    int n) {
  int t = blockIdx.x * blockDim.x + threadIdx.x;
  if (t >= n) return;
  int r = t >> 8, c = t & 255;
  int id = ids[r];
  id = id < 0 ? 0 : (id > embRows - 1 ? embRows - 1 : id);
  float v = emb[(size_t)id * HID + c];
  xf[t] = v;
  xb[t] = (__bf16)v;
}
__global__ void __launch_bounds__(256)
classify_kernel(const float* __restrict__ X, const float* __restrict__ Wc,
                const float* __restrict__ bc, float* __restrict__ out, int M) {
  int wid = threadIdx.x >> 5, lane = threadIdx.x & 31;
  int row = blockIdx.x * 8 + wid;
  if (row >= M) return;
  const float* xr = X + (size_t)row * HID;
  float acc = 0.f;
#pragma unroll
  for (int j = 0; j < 8; ++j) acc += xr[lane + 32 * j] * Wc[lane + 32 * j];
#pragma unroll
  for (int off = 16; off > 0; off >>= 1) acc += __shfl_xor(acc, off, 32);
  if (lane == 0) out[row] = acc + bc[0];
}

// ---------------------------------------------------------------------------
// Host orchestration
// ---------------------------------------------------------------------------
extern "C" void kernel_launch(void* const* d_in, const int* in_sizes, int n_in,
                              void* d_out, int out_size, void* d_ws, size_t ws_size,
                              hipStream_t stream) {
  (void)in_sizes; (void)out_size; (void)ws_size;
  const int NEn = 65536, NDi = 20000, NMe = 10000, Eg = 262144;
  const int nNodes[3] = {NEn, NDi, NMe};            // internal: 0=enc,1=diag,2=med
  const size_t rowOff[3] = {0, (size_t)NEn, (size_t)NEn + NDi};
  const size_t NT = (size_t)NEn + NDi + NMe;        // 95536
  const int alphaPos[3] = {1, 0, 2};                // internal type -> alphabetical slot
  // relations in alphabetical (param) order: d2e, e2d, e2m, m2e
  const int relSrc[4] = {1, 0, 0, 2};
  const int relDst[4] = {0, 1, 2, 0};
  const int relEI[4]  = {4, 3, 5, 6};               // d_in slot of edge-index array
  size_t srcOff[4], dstOff[4], so = 0, dn = 0;
  for (int r = 0; r < 4; ++r) {
    srcOff[r] = so; so += (size_t)nNodes[relSrc[r]];
    dstOff[r] = dn; dn += (size_t)nNodes[relDst[r]];
  }
  const size_t SRCROWS = so, DSTROWS = dn;          // both 161072
  if (n_in < 91) return;

  // --- params flattened alphabetically (JAX tree_flatten order) -------------
  // 0:x_enc 1:diag_ids 2:med_ids 3:ei_e2d 4:ei_d2e 5:ei_e2m 6:ei_m2e
  // 7:b_enc 8:cls_b 9:cls_w 10:emb_diagnosis 11:emb_medication
  // 12..50 layer0, 51..89 layer1 (39 leaves each), 90:w_enc
  auto P = [&](int i) { return (const float*)d_in[i]; };
  const int LB[2] = {12, 51};
  auto p_ab   = [&](int l, int t) { return P(LB[l] + 0  + alphaPos[t]); };
  auto p_arel = [&](int l, int r) { return P(LB[l] + 3  + r); };
  auto p_aw   = [&](int l, int t) { return P(LB[l] + 7  + alphaPos[t]); };
  auto p_kb   = [&](int l, int t) { return P(LB[l] + 10 + alphaPos[t]); };
  auto p_kw   = [&](int l, int t) { return P(LB[l] + 13 + alphaPos[t]); };
  auto p_mrel = [&](int l, int r) { return P(LB[l] + 16 + r); };
  auto p_prel = [&](int l, int r) { return P(LB[l] + 20 + r); };
  auto p_qb   = [&](int l, int t) { return P(LB[l] + 24 + alphaPos[t]); };
  auto p_qw   = [&](int l, int t) { return P(LB[l] + 27 + alphaPos[t]); };
  auto p_skip = [&](int l, int t) { return P(LB[l] + 30 + alphaPos[t]); };
  auto p_vb   = [&](int l, int t) { return P(LB[l] + 33 + alphaPos[t]); };
  auto p_vw   = [&](int l, int t) { return P(LB[l] + 36 + alphaPos[t]); };

  // --- workspace carve (~690 MB) --------------------------------------------
  char* ws = (char*)d_ws;
  size_t off = 0;
  auto carve = [&](size_t bytes) -> char* {
    char* p = ws + off;
    off += (bytes + 255) & ~(size_t)255;
    return p;
  };
  float*    xf    = (float*)  carve(NT * HID * 4);
  __bf16*   xb    = (__bf16*) carve(NT * HID * 2);
  __bf16*   kb    = (__bf16*) carve(NT * HID * 2);  // reused as GELU buffer late in layer
  __bf16*   qb    = (__bf16*) carve(NT * HID * 2);
  __bf16*   vb    = (__bf16*) carve(NT * HID * 2);
  __bf16*   krel  = (__bf16*) carve(SRCROWS * HID * 2);
  __bf16*   vrel  = (__bf16*) carve(SRCROWS * HID * 2);
  float*    alph  = (float*)  carve((size_t)4 * Eg * HEADS * 4);
  unsigned* amax  = (unsigned*)carve(DSTROWS * HEADS * 4);
  float*    denom = (float*)  carve(DSTROWS * HEADS * 4);
  float*    scat  = (float*)  carve(DSTROWS * HID * 4);
  __bf16*   xencb = (__bf16*) carve((size_t)NEn * 128 * 2);
  __bf16*   wencb = (__bf16*) carve((size_t)128 * HID * 2);
  __bf16*   wlb[2];
  for (int l = 0; l < 2; ++l) wlb[l] = (__bf16*)carve((size_t)(12 * 65536 + 8 * 1024) * 2);
  // per-layer frag-packed bf16 weight layout inside wlb[l]:
  auto w_kw = [&](int l, int t) { return wlb[l] + (size_t)(0 + t) * 65536; };
  auto w_qw = [&](int l, int t) { return wlb[l] + (size_t)(3 + t) * 65536; };
  auto w_vw = [&](int l, int t) { return wlb[l] + (size_t)(6 + t) * 65536; };
  auto w_aw = [&](int l, int t) { return wlb[l] + (size_t)(9 + t) * 65536; };
  auto w_ar = [&](int l, int r) { return wlb[l] + (size_t)12 * 65536 + (size_t)r * 8192; };
  auto w_mr = [&](int l, int r) { return wlb[l] + (size_t)12 * 65536 + (size_t)(4 + r) * 8192; };

  auto swz = [&](const float* W, __bf16* dst, int K, int N, int batch) {
    int total = batch * (K >> 5) * (N >> 4) * 32;
    swizzle_b_kernel<<<(total + 255) / 256, 256, 0, stream>>>(W, dst, K, N, batch);
  };
  auto gemm = [&](const __bf16* A, int lda, const __bf16* W, const float* bias,
                  float* Cf, __bf16* Cb, int M, int K,
                  const float* Xold, const float* skipPtr) {
    dim3 g(M / 16, 4);  // N = 256 -> 4 blocks of 64 cols
    gemm_bf16_kernel<<<g, 128, 0, stream>>>(A, lda, W, bias, Cf, Cb, M, HID, K, Xold, skipPtr);
  };

  // --- stage 0: weight swizzles + input conversion + embeddings -------------
  f32_to_bf16_kernel<<<(NEn * 128) / 256, 256, 0, stream>>>(P(0), xencb, NEn * 128);
  swz(P(90), wencb, 128, HID, 1);
  for (int l = 0; l < 2; ++l) {
    for (int t = 0; t < 3; ++t) {
      swz(p_kw(l, t), w_kw(l, t), HID, HID, 1);
      swz(p_qw(l, t), w_qw(l, t), HID, HID, 1);
      swz(p_vw(l, t), w_vw(l, t), HID, HID, 1);
      swz(p_aw(l, t), w_aw(l, t), HID, HID, 1);
    }
    for (int r = 0; r < 4; ++r) {
      swz(p_arel(l, r), w_ar(l, r), DD, DD, HEADS);
      swz(p_mrel(l, r), w_mr(l, r), DD, DD, HEADS);
    }
  }
  // x[encounter] = x_enc @ w_enc + b_enc
  gemm(xencb, 128, wencb, P(7), xf, xb, NEn, 128, nullptr, nullptr);
  // x[diagnosis], x[medication] = embedding gathers
  gather_embed_kernel<<<(NDi * HID) / 256, 256, 0, stream>>>(
      (const int*)d_in[1], P(10), NDi, xf + rowOff[1] * HID, xb + rowOff[1] * HID, NDi * HID);
  gather_embed_kernel<<<(NMe * HID) / 256, 256, 0, stream>>>(
      (const int*)d_in[2], P(11), NMe, xf + rowOff[2] * HID, xb + rowOff[2] * HID, NMe * HID);

  // --- layers ---------------------------------------------------------------
  const int EH = Eg * HEADS;  // 2,097,152 (exact multiple of 256)
  for (int l = 0; l < 2; ++l) {
    // K/Q/V projections (bf16 out)
    for (int t = 0; t < 3; ++t) {
      gemm(xb + rowOff[t] * HID, HID, w_kw(l, t), p_kb(l, t), nullptr,
           kb + rowOff[t] * HID, nNodes[t], HID, nullptr, nullptr);
      gemm(xb + rowOff[t] * HID, HID, w_qw(l, t), p_qb(l, t), nullptr,
           qb + rowOff[t] * HID, nNodes[t], HID, nullptr, nullptr);
      gemm(xb + rowOff[t] * HID, HID, w_vw(l, t), p_vb(l, t), nullptr,
           vb + rowOff[t] * HID, nNodes[t], HID, nullptr, nullptr);
    }
    // per-head relation transforms (WMMA, K=32)
    for (int r = 0; r < 4; ++r) {
      int st = relSrc[r];
      int waves = (nNodes[st] / 16) * HEADS;
      int blocks = (waves + 3) / 4;
      rel_transform_kernel<<<blocks, 128, 0, stream>>>(
          kb + rowOff[st] * HID, w_ar(l, r), krel + srcOff[r] * HID, nNodes[st]);
      rel_transform_kernel<<<blocks, 128, 0, stream>>>(
          vb + rowOff[st] * HID, w_mr(l, r), vrel + srcOff[r] * HID, nNodes[st]);
    }
    // init segment buffers: amax = ord(-inf), denom = 0, scat = 0
    {
      int n1 = (int)(DSTROWS * HEADS);
      fill_u32_kernel<<<(n1 + 255) / 256, 256, 0, stream>>>(amax, 0x007FFFFFu, n1);
      fill_u32_kernel<<<(n1 + 255) / 256, 256, 0, stream>>>((unsigned*)denom, 0u, n1);
      int n2 = (int)(DSTROWS * HID);
      fill_u32_kernel<<<(n2 + 255) / 256, 256, 0, stream>>>((unsigned*)scat, 0u, n2);
    }
    // attention logits + segment max
    for (int r = 0; r < 4; ++r) {
      const int* ei = (const int*)d_in[relEI[r]];
      edge_alpha_kernel<<<EH / 256, 256, 0, stream>>>(
          ei, ei + Eg, qb + rowOff[relDst[r]] * HID, krel + srcOff[r] * HID,
          p_prel(l, r), alph + (size_t)r * EH, amax + dstOff[r] * HEADS, EH);
    }
    {
      int n1 = (int)(DSTROWS * HEADS);
      amax_fin_kernel<<<(n1 + 255) / 256, 256, 0, stream>>>(amax, n1);
    }
    // exp + segment sum, then weighted scatter of v_rel
    for (int r = 0; r < 4; ++r) {
      const int* ei = (const int*)d_in[relEI[r]];
      edge_expsum_kernel<<<EH / 256, 256, 0, stream>>>(
          ei + Eg, alph + (size_t)r * EH, (const float*)amax + dstOff[r] * HEADS,
          denom + dstOff[r] * HEADS, EH);
    }
    for (int r = 0; r < 4; ++r) {
      const int* ei = (const int*)d_in[relEI[r]];
      edge_scatter_kernel<<<EH / 256, 256, 0, stream>>>(
          ei, ei + Eg, alph + (size_t)r * EH, vrel + srcOff[r] * HID,
          scat + dstOff[r] * HID, EH);
    }
    // combine per destination type -> GELU -> bf16 (reuse kb as gelu buffer)
    // enc <- d2e (r0) + m2e (r3); diag <- e2d (r1); med <- e2m (r2)
    combine_gelu_kernel<<<(NEn * HID) / 256, 256, 0, stream>>>(
        scat + dstOff[0] * HID, denom + dstOff[0] * HEADS,
        scat + dstOff[3] * HID, denom + dstOff[3] * HEADS,
        kb + rowOff[0] * HID, NEn * HID);
    combine_gelu_kernel<<<(NDi * HID) / 256, 256, 0, stream>>>(
        scat + dstOff[1] * HID, denom + dstOff[1] * HEADS, nullptr, nullptr,
        kb + rowOff[1] * HID, NDi * HID);
    combine_gelu_kernel<<<(NMe * HID) / 256, 256, 0, stream>>>(
        scat + dstOff[2] * HID, denom + dstOff[2] * HEADS, nullptr, nullptr,
        kb + rowOff[2] * HID, NMe * HID);
    // output projection + sigmoid-gated skip + ReLU (in-place on xf/xb)
    for (int t = 0; t < 3; ++t) {
      gemm(kb + rowOff[t] * HID, HID, w_aw(l, t), p_ab(l, t),
           xf + rowOff[t] * HID, xb + rowOff[t] * HID, nNodes[t], HID,
           xf + rowOff[t] * HID, p_skip(l, t));
    }
  }

  // --- classifier: logits = x[encounter] @ cls_w + cls_b --------------------
  classify_kernel<<<NEn / 8, 256, 0, stream>>>(xf, P(9), P(8), (float*)d_out, NEn);
}
